// GraphCIW_52132313039066
// MI455X (gfx1250) — matrix operand order repaired and checked by reading, not compiled, not run on hardware
//
#include <hip/hip_runtime.h>
#include <hip/hip_bf16.h>
#include <math.h>

#define NN 10000
#define EE 320000

typedef __attribute__((ext_vector_type(16))) _Float16 v16h;
typedef __attribute__((ext_vector_type(8)))  float    v8f;

__device__ __forceinline__ float gelu_f(float x){
  return 0.5f * x * (1.0f + erff(x * 0.70710678118654752440f));
}

// ---- WMMA fragment helpers (gfx1250 wave32 16x16x32 f16 layouts, ISA 7.12.2) ----
__device__ __forceinline__ v16h frag_a16(const _Float16* A, int lda, int lane){
  int m  = lane & 15;
  int kb = (lane & 16) ? 8 : 0;
  v16h a;
#pragma unroll
  for (int i = 0; i < 8; ++i){
    a[i]     = A[m * lda + kb + i];        // K = kb..kb+7
    a[8 + i] = A[m * lda + 16 + kb + i];   // K = 16+kb..16+kb+7
  }
  return a;
}
__device__ __forceinline__ v16h frag_b16(const _Float16* B, int ldb, int lane){
  int n  = lane & 15;
  int kb = (lane & 16) ? 8 : 0;
  v16h b;
#pragma unroll
  for (int i = 0; i < 8; ++i){ b[i] = B[(kb + i) * ldb + n]; b[8 + i] = B[(16 + kb + i) * ldb + n]; }
  return b;
}
__device__ __forceinline__ void store_c16(v8f c, float* C, int ldc, int lane){
  int n  = lane & 15;
  int rb = (lane & 16) ? 8 : 0;
#pragma unroll
  for (int j = 0; j < 8; ++j) C[(rb + j) * ldc + n] = c[j];
}

// ------------------------------- GNN front-end --------------------------------
__global__ void k_zero(float* p, long n){
  long i = (long)blockIdx.x * blockDim.x + threadIdx.x;
  if (i < n) p[i] = 0.f;
}

__global__ void k_deg(const int* ei, float* deg){
  int e = blockIdx.x * blockDim.x + threadIdx.x;
  if (e < EE) atomicAdd(&deg[ei[EE + e]], 1.f);
}

__global__ void k_dis(const float* deg, float* dis){
  int i = blockIdx.x * blockDim.x + threadIdx.x;
  if (i < NN){
    float d = deg[i];
    dis[i] = (d > 0.f) ? rsqrtf(fmaxf(d, 1.f)) : 0.f;
  }
}

__global__ void k_prelin(const float* x, const float* w, const float* b, float* z){
  int i = blockIdx.x * blockDim.x + threadIdx.x;
  if (i >= NN * 64) return;
  int n = i >> 6, d = i & 63;
  const float* xr = x + (size_t)n * 128;
  const float* wr = w + d * 128;
  float acc = b[d];
#pragma unroll 4
  for (int k = 0; k < 128; ++k) acc += xr[k] * wr[k];
  z[i] = acc;
}

__global__ void k_scatter(const int* ei, const float* dis, const float* z, float* hi){
  long i = (long)blockIdx.x * blockDim.x + threadIdx.x;
  if (i >= (long)EE * 64) return;
  int e = (int)(i >> 6), d = (int)(i & 63);
  int r = ei[e], c = ei[EE + e];
  float nm = dis[r] * dis[c];
  atomicAdd(&hi[(size_t)c * 64 + d], nm * z[(size_t)r * 64 + d]);
}

__global__ void k_conv(const float* hi, const float* z, const float* W, float* outz){
  int i = blockIdx.x * blockDim.x + threadIdx.x;
  if (i >= NN * 64) return;
  int n = i >> 6, d = i & 63;
  const float* hr = hi + (size_t)n * 64;
  const float* zr = z  + (size_t)n * 64;
  const float* wr = W + d * 128;
  float acc = 0.f;
#pragma unroll 4
  for (int k = 0; k < 64; ++k) acc += hr[k] * wr[k];
#pragma unroll 4
  for (int k = 0; k < 64; ++k) acc += zr[k] * wr[64 + k];
  outz[i] = gelu_f(acc);
}

// ---------------- precompute constants (label row is node-invariant) ----------
__global__ void k_precomp(const float* label, const float* g1, const float* b1,
                          const float* wq, const float* bq,
                          const float* wk, const float* bk,
                          const float* wv, const float* bv,
                          const float* wo, const float* bo,
                          float* cvec, float* dvec, float* Mmat,
                          _Float16* xn64h, _Float16* Pg){
  __shared__ float lstat[2];
  __shared__ float labn[64];
  __shared__ float qv[64];
  int tid = threadIdx.x;
  if (tid == 0){
    float m = 0.f; for (int k = 0; k < 64; ++k) m += label[k]; m *= (1.f/64.f);
    float v = 0.f; for (int k = 0; k < 64; ++k){ float d = label[k]-m; v += d*d; } v *= (1.f/64.f);
    lstat[0] = m; lstat[1] = rsqrtf(v + 1e-5f);
  }
  __syncthreads();
  if (tid < 64){
    float y = (label[tid] - lstat[0]) * lstat[1] * g1[tid] + b1[tid];
    labn[tid] = y; xn64h[tid] = (_Float16)y;
  }
  __syncthreads();
  if (tid < 64){
    float acc = bq[tid];
    for (int k = 0; k < 64; ++k) acc += labn[k] * wq[tid*64 + k];
    qv[tid] = acc;
  }
  for (int i = tid; i < 64*16; i += 256) Pg[i] = (_Float16)0.f;
  __syncthreads();
  { // P[k*16+h] = sum_j qv[16h+j] * wk[(16h+j)*64+k]   (256 threads exactly)
    int k = tid >> 2, h = tid & 3;
    float acc = 0.f;
    for (int j = 0; j < 16; ++j) acc += qv[h*16 + j] * wk[(h*16 + j)*64 + k];
    Pg[k*16 + h] = (_Float16)acc;
  }
  if (tid < 4){
    float acc = 0.f;
    for (int j = 0; j < 16; ++j) acc += qv[tid*16 + j] * bk[tid*16 + j];
    cvec[tid] = acc;
  }
  if (tid < 64){
    float acc = bo[tid];
    for (int j = 0; j < 64; ++j) acc += wo[tid*64 + j] * bv[j];
    dvec[tid] = acc;
  }
  for (int t = tid; t < 4*64*64; t += 256){ // M[h][o][k] = wo[o,16h+j] wv[16h+j,k]
    int h = t >> 12, o = (t >> 6) & 63, k = t & 63;
    float acc = 0.f;
    for (int j = 0; j < 16; ++j) acc += wo[o*64 + h*16 + j] * wv[(h*16 + j)*64 + k];
    Mmat[t] = acc;
  }
}

// ---------------- fused per-node transformer (one block per node) -------------
__global__ __launch_bounds__(128)
void k_node(const float* z2, const float* fe, const float* label,
            const float* g1, const float* b1, const float* g2, const float* b2,
            const float* Amat, const float* alpha_p,
            const float* w1, const float* fb1, const float* w2, const float* fb2,
            const float* cw, const float* cb,
            const float* cvec, const float* dvec, const float* Mmat,
            const _Float16* xn64h, const _Float16* Pg,
            float* out){
  __shared__ _Float16 Xn[96 * 64];     // LN1 tokens, rows 65..95 zero padded
  __shared__ _Float16 attnM[16 * 96];  // rows 0..3 = per-head attn probs
  __shared__ float Sbuf[80 * 16];      // raw scores per head column
  __shared__ float Sagg[16 * 64];      // s_h aggregates
  __shared__ float zrow[64];
  __shared__ float st[2];
  __shared__ float X1[64];
  __shared__ float HLN[64];
  __shared__ float T1[128];
  __shared__ float X2[64];

  const int n    = blockIdx.x;
  const int tid  = threadIdx.x;
  const int wv   = tid >> 5;
  const int lane = tid & 31;

  if (tid < 64) zrow[tid] = z2[(size_t)n * 64 + tid];
  __syncthreads();
  if (tid == 0){
    float m = 0.f; for (int k = 0; k < 64; ++k) m += zrow[k]; m *= (1.f/64.f);
    float v = 0.f; for (int k = 0; k < 64; ++k){ float d = zrow[k]-m; v += d*d; } v *= (1.f/64.f);
    st[0] = m; st[1] = rsqrtf(v + 1e-5f);
  }
  __syncthreads();

  // Build Xn = LN1(Zn[s] * feat_emb[s,:]) rows; row 64 = const; rows 65..95 = 0
  if (tid < 64){
    int s = tid;
    float zn = (zrow[s] - st[0]) * st[1];
    const float* fr = fe + s * 64;
    float m2 = 0.f;
    for (int d = 0; d < 64; ++d) m2 += zn * fr[d];
    m2 *= (1.f/64.f);
    float var = 0.f;
    for (int d = 0; d < 64; ++d){ float t = zn * fr[d] - m2; var += t * t; }
    float rv = rsqrtf(var * (1.f/64.f) + 1e-5f);
    for (int d = 0; d < 64; ++d)
      Xn[s*64 + d] = (_Float16)((zn * fr[d] - m2) * rv * g1[d] + b1[d]);
  } else {
    int s = tid;                       // 64..127
    if (s == 64){
      for (int d = 0; d < 64; ++d) Xn[64*64 + d] = xn64h[d];
    } else if (s < 96){
      for (int d = 0; d < 64; ++d) Xn[s*64 + d] = (_Float16)0.f;
    }
  }
  __syncthreads();

  // scores staging: Sbuf(80x16) = Xn(80x64) @ P(64x16)  [WMMA]
  for (int mi = wv; mi < 5; mi += 4){
    v8f c = {0.f,0.f,0.f,0.f,0.f,0.f,0.f,0.f};
#pragma unroll
    for (int kk = 0; kk < 2; ++kk){
      v16h a = frag_a16(&Xn[mi*16*64 + kk*32], 64, lane);
      v16h b = frag_b16(Pg + kk*32*16, 16, lane);
      c = __builtin_amdgcn_wmma_f32_16x16x32_f16(false, a, false, b, (short)0, c, false, false);
    }
    store_c16(c, &Sbuf[mi*16*16], 16, lane);
  }
  __syncthreads();

  for (int i = tid; i < 16*96; i += 128) attnM[i] = (_Float16)0.f;
  __syncthreads();

  // per-head softmax over t=0..64 (threads 0..3)
  if (tid < 4){
    int h = tid;
    float alpha = alpha_p[0];
    float ch = cvec[h];
    float mx = -3.0e38f;
    for (int t = 0; t < 65; ++t){
      float v = (Sbuf[t*16 + h] + ch) * 0.25f;
      if (t < 64) v += alpha * Amat[t*65 + 64];
      mx = fmaxf(mx, v);
    }
    float ssum = 0.f;
    for (int t = 0; t < 65; ++t){
      float v = (Sbuf[t*16 + h] + ch) * 0.25f;
      if (t < 64) v += alpha * Amat[t*65 + 64];
      float e = expf(v - mx);
      ssum += e;
      Sbuf[t*16 + h] = e;
    }
    float inv = 1.f / ssum;
    for (int t = 0; t < 65; ++t) attnM[h*96 + t] = (_Float16)(Sbuf[t*16 + h] * inv);
  }
  __syncthreads();

  // aggregation: Sagg(16x64) = attnM(16x96) @ Xn(96x64)  [WMMA, one tile/wave]
  {
    int ni = wv;
    v8f c = {0.f,0.f,0.f,0.f,0.f,0.f,0.f,0.f};
#pragma unroll
    for (int kk = 0; kk < 3; ++kk){
      v16h a = frag_a16(&attnM[kk*32], 96, lane);
      v16h b = frag_b16(&Xn[kk*32*64 + ni*16], 64, lane);
      c = __builtin_amdgcn_wmma_f32_16x16x32_f16(false, a, false, b, (short)0, c, false, false);
    }
    store_c16(c, &Sagg[ni*16], 64, lane);
  }
  __syncthreads();

  // X1[64-row] = label + sum_h M_h @ s_h + (wo@bv + bo)
  if (tid < 64){
    int o = tid;
    float acc = dvec[o];
#pragma unroll
    for (int h = 0; h < 4; ++h){
      const float* Mr = Mmat + h*4096 + o*64;
      const float* sr = Sagg + h*64;
      for (int k = 0; k < 64; ++k) acc += Mr[k] * sr[k];
    }
    X1[o] = label[o] + acc;
  }
  __syncthreads();
  if (tid == 0){
    float m = 0.f; for (int k = 0; k < 64; ++k) m += X1[k]; m *= (1.f/64.f);
    float v = 0.f; for (int k = 0; k < 64; ++k){ float d = X1[k]-m; v += d*d; } v *= (1.f/64.f);
    st[0] = m; st[1] = rsqrtf(v + 1e-5f);
  }
  __syncthreads();
  if (tid < 64) HLN[tid] = (X1[tid] - st[0]) * st[1] * g2[tid] + b2[tid];
  __syncthreads();
  { // FFN up: 128 threads, one col each
    int j = tid;
    float acc = fb1[j];
    for (int k = 0; k < 64; ++k) acc += HLN[k] * w1[j*64 + k];
    T1[j] = gelu_f(acc);
  }
  __syncthreads();
  if (tid < 64){
    int o = tid;
    float acc = fb2[o];
    for (int j = 0; j < 128; ++j) acc += T1[j] * w2[o*128 + j];
    X2[o] = X1[o] + acc;
  }
  __syncthreads();
  if (tid < 10){
    float acc = cb[tid];
    for (int k = 0; k < 64; ++k) acc += X2[k] * cw[tid*64 + k];
    out[(size_t)n*10 + tid] = acc;
  }
}

// ---------------------------------- launch ------------------------------------
extern "C" void kernel_launch(void* const* d_in, const int* in_sizes, int n_in,
                              void* d_out, int out_size, void* d_ws, size_t ws_size,
                              hipStream_t stream){
  const float* x     = (const float*)d_in[0];
  const int*   ei    = (const int*)  d_in[1];
  const float* pre_w = (const float*)d_in[2];
  const float* pre_b = (const float*)d_in[3];
  const float* c1w   = (const float*)d_in[4];
  const float* c2w   = (const float*)d_in[5];
  const float* fe    = (const float*)d_in[6];
  const float* label = (const float*)d_in[7];
  const float* wq    = (const float*)d_in[8];
  const float* bq    = (const float*)d_in[9];
  const float* wk    = (const float*)d_in[10];
  const float* bk    = (const float*)d_in[11];
  const float* wvv   = (const float*)d_in[12];
  const float* bv    = (const float*)d_in[13];
  const float* wo    = (const float*)d_in[14];
  const float* bo    = (const float*)d_in[15];
  const float* alpha = (const float*)d_in[16];
  const float* w1    = (const float*)d_in[17];
  const float* fb1   = (const float*)d_in[18];
  const float* w2    = (const float*)d_in[19];
  const float* fb2   = (const float*)d_in[20];
  const float* g1    = (const float*)d_in[21];
  const float* b1    = (const float*)d_in[22];
  const float* g2    = (const float*)d_in[23];
  const float* b2    = (const float*)d_in[24];
  const float* Amat  = (const float*)d_in[25];
  const float* cw    = (const float*)d_in[26];
  const float* cb    = (const float*)d_in[27];
  float* out = (float*)d_out;

  float* deg  = (float*)d_ws;
  float* dis  = deg  + NN;
  float* bufA = dis  + NN;
  float* bufB = bufA + (size_t)NN * 64;
  float* bufC = bufB + (size_t)NN * 64;
  float* cvec = bufC + (size_t)NN * 64;
  float* dvec = cvec + 4;
  float* Mmat = dvec + 64;
  _Float16* xn64h = (_Float16*)(Mmat + 4*64*64);
  _Float16* Pg    = xn64h + 64;

  const long NE64 = (long)EE * 64;
  const long ND   = (long)NN * 64;

  // degree + norm
  k_zero<<<(NN + 255)/256, 256, 0, stream>>>(deg, NN);
  k_deg <<<(EE + 255)/256, 256, 0, stream>>>(ei, deg);
  k_dis <<<(NN + 255)/256, 256, 0, stream>>>(deg, dis);

  // pre-linear
  k_prelin<<<(int)((ND + 255)/256), 256, 0, stream>>>(x, pre_w, pre_b, bufA);

  // conv1
  k_zero   <<<(int)((ND + 255)/256),   256, 0, stream>>>(bufB, ND);
  k_scatter<<<(int)((NE64 + 255)/256), 256, 0, stream>>>(ei, dis, bufA, bufB);
  k_conv   <<<(int)((ND + 255)/256),   256, 0, stream>>>(bufB, bufA, c1w, bufC);

  // conv2
  k_zero   <<<(int)((ND + 255)/256),   256, 0, stream>>>(bufB, ND);
  k_scatter<<<(int)((NE64 + 255)/256), 256, 0, stream>>>(ei, dis, bufC, bufB);
  k_conv   <<<(int)((ND + 255)/256),   256, 0, stream>>>(bufB, bufC, c2w, bufA);

  // node-invariant transformer constants
  k_precomp<<<1, 256, 0, stream>>>(label, g1, b1, wq, bq, wk, bk, wvv, bv, wo, bo,
                                   cvec, dvec, Mmat, xn64h, Pg);

  // fused per-node transformer + classifier
  k_node<<<NN, 128, 0, stream>>>(bufA, fe, label, g1, b1, g2, b2, Amat, alpha,
                                 w1, fb1, w2, fb2, cw, cb,
                                 cvec, dvec, Mmat, xn64h, Pg, out);
}
